// ChipPairEncoderAlpha_1382979469738
// MI455X (gfx1250) — matrix operand order repaired and checked by reading, not compiled, not run on hardware
//
#include <hip/hip_runtime.h>
#include <hip/hip_bf16.h>

typedef __attribute__((ext_vector_type(16))) _Float16 v16h;
typedef __attribute__((ext_vector_type(8)))  float    v8f;
typedef __attribute__((ext_vector_type(2)))  float    v2f;

// ---------- helpers ----------
__device__ __forceinline__ float wave_sum32(float v) {
  #pragma unroll
  for (int o = 16; o > 0; o >>= 1) v += __shfl_xor(v, o, 32);
  return v;
}

// f16 WMMA operand index maps (CDNA5 ISA §7.12.2)
__device__ __forceinline__ int kA16(int q, int hi) { return (q < 8 ? q : q + 8) + (hi ? 8 : 0); }
__device__ __forceinline__ int kB16(int q, int hi) { return (q < 8 ? q : q + 8) + (hi ? 8 : 0); }

// CDNA5 async global->LDS copy (ASYNCcnt path, CDNA5 ISA op 98)
__device__ __forceinline__ void async_load_b128(const void* g, void* l) {
  unsigned lds_off = (unsigned)(unsigned long long)l;   // low 32 bits = LDS offset
  asm volatile("global_load_async_to_lds_b128 %0, %1, off"
               :: "v"(lds_off), "v"(g) : "memory");
}
__device__ __forceinline__ void wait_async0() {
  asm volatile("s_wait_asynccnt 0x0" ::: "memory");
}

// ---------- K1: build x = LN(msa + pos)  -> ws_x (B,S,L,C) ----------
__global__ __launch_bounds__(256) void k1_build_x(
    const float* __restrict__ ctcf, const float* __restrict__ hac,
    const float* __restrict__ me1,  const float* __restrict__ me3,
    const float* __restrict__ we_w, const float* __restrict__ we_b,
    const float* __restrict__ pos,  const float* __restrict__ mn_g,
    const float* __restrict__ mn_b, float* __restrict__ xw)
{
  int wave = threadIdx.x >> 5, lane = threadIdx.x & 31;
  int r = blockIdx.x * 8 + wave;              // row index in [0, B*S*L)
  int b = r >> 8, rem = r & 255, s = rem >> 6, l = rem & 63;
  const float* sig = (s == 0) ? ctcf : (s == 1) ? hac : (s == 2) ? me1 : me3;
  float sv = sig[b * 64 + l];
  float v = sv * we_w[lane] + we_b[lane] + pos[l * 32 + lane];
  float m = wave_sum32(v) * 0.03125f;
  float d = v - m;
  float var = wave_sum32(d * d) * 0.03125f;
  xw[(size_t)r * 32 + lane] = d * rsqrtf(var + 1e-5f) * mn_g[lane] + mn_b[lane];
}

// ---------- K2: fused row attention per (b,s); x updated in place ----------
__global__ __launch_bounds__(256) void k2_attn(
    float* __restrict__ xw, const float* __restrict__ bulk,
    const float* __restrict__ wq, const float* __restrict__ bq,
    const float* __restrict__ wk, const float* __restrict__ bk,
    const float* __restrict__ wv, const float* __restrict__ bv,
    const float* __restrict__ wo, const float* __restrict__ bo,
    const float* __restrict__ an_g, const float* __restrict__ an_b,
    const float* __restrict__ conv_w, const float* __restrict__ conv_b,
    const float* __restrict__ gate_w, const float* __restrict__ gate_b)
{
  __shared__ float    xs[64][32];
  __shared__ _Float16 xh[64][32];
  __shared__ float    qf[64][32], kf[64][32], vf[64][32];
  __shared__ float    sc[4][64][64];          // scores -> attn per head
  __shared__ float    og[64][32];             // gated AV output (l, h*8+d)
  __shared__ float    gt[64][32];             // gates, then out-projection
  __shared__ _Float16 oh[64][32];

  const int tid = threadIdx.x, wave = tid >> 5, lane = tid & 31;
  const int lm = lane & 15; const int hi = (lane >= 16) ? 1 : 0;
  const int bs = blockIdx.x, b = bs >> 2;
  float* xg = xw + (size_t)bs * 2048;

  // async global->LDS copy of the f32 x tile (2 x b128 per thread)
  for (int i = tid * 4; i < 2048; i += 1024)
    async_load_b128(xg + i, &xs[0][0] + i);
  wait_async0();
  __syncthreads();
  // derive f16 copy from LDS
  for (int i = tid; i < 2048; i += 256)
    xh[i >> 5][i & 31] = (_Float16)xs[i >> 5][i & 31];
  __syncthreads();

  // ---- Q/K/V projections: x(64x32) @ W^T(32x32), 24 tiles over 8 waves ----
  for (int t = wave; t < 24; t += 8) {
    int g = t >> 3, ti = t & 7, mt = ti >> 1, nt = ti & 1;
    const float* W    = (g == 0) ? wq : (g == 1) ? wk : wv;
    const float* bias = (g == 0) ? bq : (g == 1) ? bk : bv;
    v16h a, bm;
    #pragma unroll
    for (int q = 0; q < 16; ++q) {
      a[q]  = xh[mt * 16 + lm][kA16(q, hi)];
      bm[q] = (_Float16)W[(nt * 16 + lm) * 32 + kB16(q, hi)];   // B[k][n] = W[n][k]
    }
    v8f c = {};
    c = __builtin_amdgcn_wmma_f32_16x16x32_f16(false, a, false, bm, (short)0, c, false, false);
    float* dst = (g == 0) ? &qf[0][0] : (g == 1) ? &kf[0][0] : &vf[0][0];
    int col = nt * 16 + lm;
    #pragma unroll
    for (int r = 0; r < 8; ++r)
      dst[(mt * 16 + r + hi * 8) * 32 + col] = c[r] + bias[col];
  }
  __syncthreads();

  // ---- scores per head: q.k (K=8) with fp32 WMMA 16x16x4, + bulk bias ----
  const float rsD = 0.35355339059327373f;     // 1/sqrt(8)
  const float* bmp = bulk + (size_t)b * 4096;
  for (int t = wave; t < 64; t += 8) {
    int h = t >> 4, ti = t & 15, mt = ti >> 2, nt = ti & 3;
    v8f c = {};
    #pragma unroll
    for (int ko = 0; ko < 8; ko += 4) {
      v2f a, bb;
      int k0 = h * 8 + ko + hi * 2;
      a.x  = qf[mt * 16 + lm][k0];   a.y  = qf[mt * 16 + lm][k0 + 1];
      bb.x = kf[nt * 16 + lm][k0];   bb.y = kf[nt * 16 + lm][k0 + 1];
      c = __builtin_amdgcn_wmma_f32_16x16x4_f32(false, a, false, bb, (short)0, c, false, false);
    }
    float cw = conv_w[h], cb = conv_b[h];
    int col = nt * 16 + lm;
    #pragma unroll
    for (int r = 0; r < 8; ++r) {
      int row = mt * 16 + r + hi * 8;
      sc[h][row][col] = c[r] * rsD + bmp[row * 64 + col] * cw + cb;
    }
  }
  __syncthreads();

  // ---- softmax: one thread per (h,row) ----
  {
    int h = tid >> 6, row = tid & 63;
    float mx = -1e30f;
    for (int j = 0; j < 64; ++j) mx = fmaxf(mx, sc[h][row][j]);
    float sum = 0.f;
    for (int j = 0; j < 64; ++j) { float e = __expf(sc[h][row][j] - mx); sc[h][row][j] = e; sum += e; }
    float inv = 1.0f / sum;
    for (int j = 0; j < 64; ++j) sc[h][row][j] *= inv;
  }
  __syncthreads();

  // ---- attn @ v per head (K=64, N=8 padded to 16), fp32 WMMA ----
  for (int t = wave; t < 16; t += 8) {
    int h = t >> 2, mt = t & 3;
    v8f c = {};
    for (int ko = 0; ko < 64; ko += 4) {
      v2f a, bb;
      int ka = ko + hi * 2;
      a.x  = sc[h][mt * 16 + lm][ka];  a.y = sc[h][mt * 16 + lm][ka + 1];
      bb.x = (lm < 8) ? vf[ka][h * 8 + lm] : 0.0f;
      bb.y = (lm < 8) ? vf[ka + 1][h * 8 + lm] : 0.0f;
      c = __builtin_amdgcn_wmma_f32_16x16x4_f32(false, a, false, bb, (short)0, c, false, false);
    }
    if (lm < 8) {
      #pragma unroll
      for (int r = 0; r < 8; ++r) og[mt * 16 + r + hi * 8][h * 8 + lm] = c[r];
    }
  }
  __syncthreads();

  // ---- gating: gate[l,h,c] = sigmoid(sum_d gate_w[h,c,d]*out[l,h,d] + gate_b) ----
  for (int idx = tid; idx < 2048; idx += 256) {
    int l = idx >> 5, hc = idx & 31, h = hc >> 3, cc = hc & 7;
    float acc = gate_b[h * 8 + cc];
    #pragma unroll
    for (int d = 0; d < 8; ++d) acc += gate_w[(h * 8 + cc) * 8 + d] * og[l][h * 8 + d];
    gt[l][hc] = 1.0f / (1.0f + __expf(-acc));
  }
  __syncthreads();
  for (int idx = tid; idx < 2048; idx += 256) {
    int l = idx >> 5, hc = idx & 31;
    float v = og[l][hc] * gt[l][hc];
    og[l][hc] = v;
    oh[l][hc] = (_Float16)v;
  }
  __syncthreads();

  // ---- output projection: og(64x32) @ wo^T + bo (8 tiles, 1 per wave) ----
  {
    int mt = wave >> 1, nt = wave & 1;
    v16h a, bm;
    #pragma unroll
    for (int q = 0; q < 16; ++q) {
      a[q]  = oh[mt * 16 + lm][kA16(q, hi)];
      bm[q] = (_Float16)wo[(nt * 16 + lm) * 32 + kB16(q, hi)];
    }
    v8f c = {};
    c = __builtin_amdgcn_wmma_f32_16x16x32_f16(false, a, false, bm, (short)0, c, false, false);
    int col = nt * 16 + lm;
    #pragma unroll
    for (int r = 0; r < 8; ++r)
      gt[mt * 16 + r + hi * 8][col] = c[r] + bo[col];     // gt reused as out2
  }
  __syncthreads();

  // ---- residual + post-attn LayerNorm, write back in place ----
  for (int row = wave; row < 64; row += 8) {
    float v = xs[row][lane] + gt[row][lane];
    float m = wave_sum32(v) * 0.03125f;
    float d = v - m;
    float var = wave_sum32(d * d) * 0.03125f;
    xg[row * 32 + lane] = d * rsqrtf(var + 1e-5f) * an_g[lane] + an_b[lane];
  }
}

// ---------- K3: m = max over S, row-normalized (norm factorizes) ----------
__global__ __launch_bounds__(256) void k3_rowmax(
    const float* __restrict__ xw, _Float16* __restrict__ mh)
{
  int wave = threadIdx.x >> 5, lane = threadIdx.x & 31;
  int r = blockIdx.x * 8 + wave;              // (b,l) in [0, B*L)
  float v = -1e30f;
  int b = r >> 6, l = r & 63;
  #pragma unroll
  for (int s = 0; s < 4; ++s) v = fmaxf(v, xw[(size_t)((b * 4 + s) * 64 + l) * 32 + lane]);
  float n2 = wave_sum32(v * v);
  float inv = 1.0f / fmaxf(sqrtf(n2), 1e-3f); // ||mi||*||mj|| clamp split per factor
  mh[(size_t)r * 32 + lane] = (_Float16)(v * inv);
}

// ---------- K4: feat_p = (m̂ A_p) m̂^T per batch, WMMA f16 ----------
__global__ __launch_bounds__(256) void k4_pair(
    const _Float16* __restrict__ mh, const float* __restrict__ ppw,
    float* __restrict__ featw)
{
  __shared__ _Float16 ms[64][32];
  __shared__ _Float16 pw[16 * 1024];
  __shared__ _Float16 th[16][64][32];
  const int tid = threadIdx.x, wave = tid >> 5, lane = tid & 31;
  const int lm = lane & 15; const int hi = (lane >= 16) ? 1 : 0;
  const int b = blockIdx.x;

  // m̂ tile is already f16 in the workspace: pure async global->LDS copy
  async_load_b128(mh + (size_t)b * 2048 + tid * 8, &ms[0][0] + tid * 8);
  // pp_w needs f32->f16 conversion: VGPR path
  for (int i = tid; i < 16384; i += 256) pw[i] = (_Float16)ppw[i];
  wait_async0();
  __syncthreads();

  // GEMM1: t_p[i][d] = sum_c m̂[i][c] * ppw[p][c*32+d]   (128 tiles)
  for (int t = wave; t < 128; t += 8) {
    int p = t >> 3, ti = t & 7, mt = ti >> 1, nt = ti & 1;
    v16h a, bm;
    #pragma unroll
    for (int q = 0; q < 16; ++q) {
      int kb = kB16(q, hi);
      a[q]  = ms[mt * 16 + lm][kA16(q, hi)];
      bm[q] = pw[p * 1024 + kb * 32 + nt * 16 + lm];
    }
    v8f c = {};
    c = __builtin_amdgcn_wmma_f32_16x16x32_f16(false, a, false, bm, (short)0, c, false, false);
    int col = nt * 16 + lm;
    #pragma unroll
    for (int r = 0; r < 8; ++r) th[p][mt * 16 + r + hi * 8][col] = (_Float16)c[r];
  }
  __syncthreads();

  // GEMM2: feat[p][i][j] = sum_d t_p[i][d] * m̂[j][d]   (256 tiles)
  for (int t = wave; t < 256; t += 8) {
    int p = t >> 4, ti = t & 15, mt = ti >> 2, nt = ti & 3;
    v16h a, bm;
    #pragma unroll
    for (int q = 0; q < 16; ++q) {
      a[q]  = th[p][mt * 16 + lm][kA16(q, hi)];
      bm[q] = ms[nt * 16 + lm][kB16(q, hi)];   // B[k=d][n=j] = m̂[j][d]
    }
    v8f c = {};
    c = __builtin_amdgcn_wmma_f32_16x16x32_f16(false, a, false, bm, (short)0, c, false, false);
    int col = nt * 16 + lm;
    size_t base = ((size_t)b * 16 + p) * 4096;
    #pragma unroll
    for (int r = 0; r < 8; ++r)
      featw[base + (size_t)(mt * 16 + r + hi * 8) * 64 + col] = c[r];
  }
}

// ---------- K5: +pp_b, AdaNorm over CP=16, SiLU, output (B,CP,L,L) ----------
__global__ __launch_bounds__(256) void k5_adanorm(
    const float* __restrict__ featw, const float* __restrict__ pp_b,
    const float* __restrict__ ada_g, const float* __restrict__ ada_b,
    const float* __restrict__ ada_alpha, float* __restrict__ out)
{
  int idx = blockIdx.x * 256 + threadIdx.x;   // (b, i*64+j) in [0, B*4096)
  int b = idx >> 12, ij = idx & 4095;
  size_t base = (size_t)b * 16 * 4096 + ij;
  float v[16]; float m = 0.f;
  #pragma unroll
  for (int p = 0; p < 16; ++p) { v[p] = featw[base + p * 4096] + pp_b[p]; m += v[p]; }
  m *= 0.0625f;
  float var = 0.f;
  #pragma unroll
  for (int p = 0; p < 16; ++p) { float d = v[p] - m; var += d * d; }
  var *= 0.0625f;
  float rs = rsqrtf(var + 1e-5f);
  float alpha = ada_alpha[0];
  #pragma unroll
  for (int p = 0; p < 16; ++p) {
    float f = v[p] + alpha * ((v[p] - m) * rs * ada_g[p] + ada_b[p]);
    out[base + p * 4096] = f / (1.0f + __expf(-f));       // SiLU
  }
}

extern "C" void kernel_launch(void* const* d_in, const int* in_sizes, int n_in,
                              void* d_out, int out_size, void* d_ws, size_t ws_size,
                              hipStream_t stream) {
  (void)in_sizes; (void)n_in; (void)out_size; (void)ws_size;
  const float* ctcf  = (const float*)d_in[0];
  const float* hac   = (const float*)d_in[1];
  const float* me1   = (const float*)d_in[2];
  const float* me3   = (const float*)d_in[3];
  const float* bulk  = (const float*)d_in[4];
  const float* we_w  = (const float*)d_in[5];
  const float* we_b  = (const float*)d_in[6];
  const float* pos   = (const float*)d_in[7];
  const float* mn_g  = (const float*)d_in[8];
  const float* mn_b  = (const float*)d_in[9];
  const float* wq    = (const float*)d_in[10];
  const float* bq    = (const float*)d_in[11];
  const float* wk    = (const float*)d_in[12];
  const float* bk    = (const float*)d_in[13];
  const float* wv    = (const float*)d_in[14];
  const float* bv    = (const float*)d_in[15];
  const float* wo    = (const float*)d_in[16];
  const float* bo    = (const float*)d_in[17];
  const float* an_g  = (const float*)d_in[18];
  const float* an_b  = (const float*)d_in[19];
  const float* conv_w= (const float*)d_in[20];
  const float* conv_b= (const float*)d_in[21];
  const float* gate_w= (const float*)d_in[22];
  const float* gate_b= (const float*)d_in[23];
  const float* pp_w  = (const float*)d_in[24];
  const float* pp_b  = (const float*)d_in[25];
  const float* ada_g = (const float*)d_in[26];
  const float* ada_b = (const float*)d_in[27];
  const float* ada_a = (const float*)d_in[28];

  float*     ws_x    = (float*)d_ws;                                    // 1 MiB
  _Float16*  ws_m    = (_Float16*)((char*)d_ws + (1u << 20));           // 128 KiB
  float*     ws_feat = (float*)((char*)d_ws + (1u << 20) + (1u << 17)); // 8 MiB

  k1_build_x<<<1024, 256, 0, stream>>>(ctcf, hac, me1, me3, we_w, we_b, pos, mn_g, mn_b, ws_x);
  k2_attn   <<<128,  256, 0, stream>>>(ws_x, bulk, wq, bq, wk, bk, wv, bv, wo, bo,
                                       an_g, an_b, conv_w, conv_b, gate_w, gate_b);
  k3_rowmax <<<256,  256, 0, stream>>>(ws_x, ws_m);
  k4_pair   <<<32,   256, 0, stream>>>(ws_m, pp_w, ws_feat);
  k5_adanorm<<<512,  256, 0, stream>>>(ws_feat, pp_b, ada_g, ada_b, ada_a, (float*)d_out);
}
